// MmbeddingsEncoder_79233556677137
// MI455X (gfx1250) — matrix operand order repaired and checked by reading, not compiled, not run on hardware
//
#include <hip/hip_runtime.h>
#include <math.h>

typedef _Float16 v16h __attribute__((ext_vector_type(16)));
typedef _Float16 v8h  __attribute__((ext_vector_type(8)));
typedef float    v8f  __attribute__((ext_vector_type(8)));
typedef float    v2f  __attribute__((ext_vector_type(2)));
typedef float    f8v  __attribute__((ext_vector_type(8)));

#define NQ   8192
#define H0C  128
#define H1C  64

// ---------------------------------------------------------------- zero scratch
__global__ void zero_ws_kernel(float* __restrict__ ws, int n) {
    int i = blockIdx.x * blockDim.x + threadIdx.x;
    int stride = gridDim.x * blockDim.x;
    for (; i < n; i += stride) ws[i] = 0.f;
}

// ------------------------------------------------- fused MLP + segment scatter
// block = 256 threads (8 waves), 128 rows per block, 16 rows per wave.
__global__ __launch_bounds__(256) void mlp_segsum_kernel(
    const float* __restrict__ X,  const float* __restrict__ y,
    const int*   __restrict__ ids0, const int* __restrict__ ids1,
    const float* __restrict__ W0, const float* __restrict__ b0,
    const float* __restrict__ W1, const float* __restrict__ b1,
    float* __restrict__ sum0, float* __restrict__ sum1,
    float* __restrict__ cnt0, float* __restrict__ cnt1, int Nrows)
{
    __shared__ _Float16 W0t[128][104];    // W0 transposed [n][k], K padded 65->96
    __shared__ _Float16 W1t[64][136];     // W1 transposed [n][k], K=128
    __shared__ _Float16 hs[8][16][136];   // per-wave h tile [row][n], K=128
    __shared__ int lids0[128];
    __shared__ int lids1[128];

    const int tid    = threadIdx.x;
    const int rowblk = blockIdx.x * 128;

    // stage weights (f32 -> f16, transposed, zero-padded K for layer 0)
    for (int idx = tid; idx < 128 * 96; idx += 256) {
        int n = idx / 96, k = idx % 96;
        W0t[n][k] = (k < 65) ? (_Float16)W0[k * 128 + n] : (_Float16)0.f;
    }
    for (int idx = tid; idx < 64 * 128; idx += 256) {
        int n = idx / 128, k = idx % 128;
        W1t[n][k] = (_Float16)W1[k * 64 + n];
    }
    if (tid < 128) {
        int r = rowblk + tid;
        int i0 = (r < Nrows) ? ids0[r] : 0;
        int i1 = (r < Nrows) ? ids1[r] : 0;
        lids0[tid] = i0;
        lids1[tid] = i1;
        if (r < Nrows) {
            atomicAdd(&cnt0[i0], 1.f);
            atomicAdd(&cnt1[i1], 1.f);
        }
    }
    __syncthreads();

    const int wave = tid >> 5;
    const int lane = tid & 31;
    const int lr   = lane & 15;      // row (A/C-M low) / col (B/C-N) selector
    const int half = lane >> 4;      // half-wave selector

    const int  grow = rowblk + wave * 16 + lr;
    const bool rv   = grow < Nrows;

    // ---- A fragments for layer 0 (rows of Xy, K padded to 96) --------------
    // ISA A-layout: half i<8 : K = 8*half + i ; i>=8 : K = 16 + 8*half + (i-8)
    v16h a0[3];
    #pragma unroll
    for (int kk = 0; kk < 2; ++kk) {
        v16h a;
        if (rv) {
            const float* xp = X + (size_t)grow * 64 + 32 * kk;
            f8v lo = *(const f8v*)(xp + 8 * half);
            f8v hi = *(const f8v*)(xp + 16 + 8 * half);
            #pragma unroll
            for (int i = 0; i < 8; ++i) { a[i] = (_Float16)lo[i]; a[8 + i] = (_Float16)hi[i]; }
        } else {
            #pragma unroll
            for (int i = 0; i < 16; ++i) a[i] = (_Float16)0.f;
        }
        a0[kk] = a;
    }
    {   // kk = 2 : K = 64..95 -> only K==64 (the y column) is non-zero
        v16h a;
        #pragma unroll
        for (int i = 0; i < 16; ++i) a[i] = (_Float16)0.f;
        if (rv && half == 0) a[0] = (_Float16)y[grow];
        a0[2] = a;
    }

    // ---- layer 0: h = relu(Xy @ W0 + b0) -----------------------------------
    #pragma unroll
    for (int j = 0; j < 8; ++j) {                 // 8 n-tiles over H0=128
        v8f acc = {};
        #pragma unroll
        for (int kk = 0; kk < 3; ++kk) {
            // ISA B-layout: half i<8 : K = 8*half + i ; i>=8 : K = 16 + 8*half + (i-8)
            const _Float16* wp = &W0t[16 * j + lr][32 * kk + 8 * half];
            v8h blo = *(const v8h*)wp;
            v8h bhi = *(const v8h*)(wp + 16);
            v16h b;
            #pragma unroll
            for (int i = 0; i < 8; ++i) { b[i] = blo[i]; b[8 + i] = bhi[i]; }
            acc = __builtin_amdgcn_wmma_f32_16x16x32_f16(false, a0[kk], false, b,
                                                         (short)0, acc, false, false);
        }
        const int   n    = 16 * j + lr;           // C-layout: N = lane&15
        const float bias = b0[n];
        #pragma unroll
        for (int r = 0; r < 8; ++r) {             // C-layout: M = r + 8*half
            float v = acc[r] + bias;
            v = v > 0.f ? v : 0.f;
            hs[wave][r + 8 * half][n] = (_Float16)v;
        }
    }
    __syncthreads();

    // ---- layer 1: z1 = relu(h @ W1 + b1), scatter into segment sums --------
    v16h a1[4];
    #pragma unroll
    for (int kk = 0; kk < 4; ++kk) {
        const _Float16* hp = &hs[wave][lr][32 * kk + 8 * half];
        v8h lo = *(const v8h*)hp;
        v8h hi = *(const v8h*)(hp + 16);
        v16h a;
        #pragma unroll
        for (int i = 0; i < 8; ++i) { a[i] = lo[i]; a[8 + i] = hi[i]; }
        a1[kk] = a;
    }

    #pragma unroll
    for (int t = 0; t < 4; ++t) {                 // 4 n-tiles over H1=64
        v8f acc = {};
        #pragma unroll
        for (int kk = 0; kk < 4; ++kk) {
            const _Float16* wp = &W1t[16 * t + lr][32 * kk + 8 * half];
            v8h blo = *(const v8h*)wp;
            v8h bhi = *(const v8h*)(wp + 16);
            v16h b;
            #pragma unroll
            for (int i = 0; i < 8; ++i) { b[i] = blo[i]; b[8 + i] = bhi[i]; }
            acc = __builtin_amdgcn_wmma_f32_16x16x32_f16(false, a1[kk], false, b,
                                                         (short)0, acc, false, false);
        }
        const int   n    = 16 * t + lr;
        const float bias = b1[n];
        #pragma unroll
        for (int r = 0; r < 8; ++r) {
            const int lrow = wave * 16 + r + 8 * half;
            const int gr   = rowblk + lrow;
            float v = acc[r] + bias;
            v = v > 0.f ? v : 0.f;
            if (gr < Nrows) {
                atomicAdd(&sum0[(size_t)lids0[lrow] * 64 + n], v);
                atomicAdd(&sum1[(size_t)lids1[lrow] * 64 + n], v);
            }
        }
    }
}

// -------------------------------------------- bucket head: fp32 WMMA 16x16x4
// block = 256 (8 waves), each wave owns 16 buckets.
__global__ __launch_bounds__(256) void head_kernel(
    const float* __restrict__ sum0, const float* __restrict__ sum1,
    const float* __restrict__ cnt0, const float* __restrict__ cnt1,
    const float* __restrict__ Wm0, const float* __restrict__ bm0,
    const float* __restrict__ Wv0, const float* __restrict__ bv0,
    const float* __restrict__ Wm1, const float* __restrict__ bm1,
    const float* __restrict__ Wv1, const float* __restrict__ bv1,
    const float* __restrict__ eps0, const float* __restrict__ eps1,
    float* __restrict__ out)
{
    const int wave = threadIdx.x >> 5;
    const int lane = threadIdx.x & 31;
    const int lr   = lane & 15;
    const int half = lane >> 4;
    const int segbase = (blockIdx.x * 8 + wave) * 16;
    const int s = segbase + lr;

    const float inv0 = 1.f / fmaxf(cnt0[s], 1.f);
    const float inv1 = 1.f / fmaxf(cnt1[s], 1.f);
    const float* s0p = sum0 + (size_t)s * 64;
    const float* s1p = sum1 + (size_t)s * 64;

    v8f aM0 = {}, aV0 = {}, aM1 = {}, aV1 = {};
    const int koff = 2 * half;          // f32 A/B layout: v0->K, v1->K+1; halves split K by 2
    #pragma unroll
    for (int kk = 0; kk < 16; ++kk) {   // K = 64 in steps of 4
        const int k = 4 * kk + koff;
        v2f A0  = { s0p[k] * inv0, s0p[k + 1] * inv0 };
        v2f A1  = { s1p[k] * inv1, s1p[k + 1] * inv1 };
        v2f Bm0 = { Wm0[k * 16 + lr], Wm0[(k + 1) * 16 + lr] };
        v2f Bv0 = { Wv0[k * 16 + lr], Wv0[(k + 1) * 16 + lr] };
        v2f Bm1 = { Wm1[k * 16 + lr], Wm1[(k + 1) * 16 + lr] };
        v2f Bv1 = { Wv1[k * 16 + lr], Wv1[(k + 1) * 16 + lr] };
        aM0 = __builtin_amdgcn_wmma_f32_16x16x4_f32(false, A0, false, Bm0, (short)0, aM0, false, false);
        aV0 = __builtin_amdgcn_wmma_f32_16x16x4_f32(false, A0, false, Bv0, (short)0, aV0, false, false);
        aM1 = __builtin_amdgcn_wmma_f32_16x16x4_f32(false, A1, false, Bm1, (short)0, aM1, false, false);
        aV1 = __builtin_amdgcn_wmma_f32_16x16x4_f32(false, A1, false, Bv1, (short)0, aV1, false, false);
    }

    const int d = lr;
    #pragma unroll
    for (int r = 0; r < 8; ++r) {
        const int q = segbase + r + 8 * half;
        float m0 = aM0[r] + bm0[d];
        float l0 = aV0[r] + bv0[d];
        float m1 = aM1[r] + bm1[d];
        float l1 = aV1[r] + bv1[d];
        float sp0 = m0 + expf(0.5f * l0) * eps0[q * 16 + d];
        float sp1 = m1 + expf(0.5f * l1) * eps1[q * 16 + d];
        out[(0 * NQ + q) * 16 + d] = m0;
        out[(1 * NQ + q) * 16 + d] = m1;
        out[(2 * NQ + q) * 16 + d] = l0;
        out[(3 * NQ + q) * 16 + d] = l1;
        out[(4 * NQ + q) * 16 + d] = sp0;
        out[(5 * NQ + q) * 16 + d] = sp1;
    }
}

// ---------------------------------------------------------------------- launch
extern "C" void kernel_launch(void* const* d_in, const int* in_sizes, int n_in,
                              void* d_out, int out_size, void* d_ws, size_t ws_size,
                              hipStream_t stream)
{
    const float* X    = (const float*)d_in[0];
    const float* y    = (const float*)d_in[1];
    const int*   ids0 = (const int*)d_in[2];
    const int*   ids1 = (const int*)d_in[3];
    const float* W0   = (const float*)d_in[4];
    const float* b0   = (const float*)d_in[5];
    const float* W1   = (const float*)d_in[6];
    const float* b1   = (const float*)d_in[7];
    const float* Wm0  = (const float*)d_in[8];
    const float* bm0  = (const float*)d_in[9];
    const float* Wv0  = (const float*)d_in[10];
    const float* bv0  = (const float*)d_in[11];
    const float* Wm1  = (const float*)d_in[12];
    const float* bm1  = (const float*)d_in[13];
    const float* Wv1  = (const float*)d_in[14];
    const float* bv1  = (const float*)d_in[15];
    const float* eps0 = (const float*)d_in[16];
    const float* eps1 = (const float*)d_in[17];
    const int Nrows = in_sizes[1];          // y has N elements

    float* ws   = (float*)d_ws;
    float* sum0 = ws;
    float* sum1 = sum0 + (size_t)NQ * 64;
    float* cnt0 = sum1 + (size_t)NQ * 64;
    float* cnt1 = cnt0 + NQ;
    const int ws_elems = 2 * NQ * 64 + 2 * NQ;

    zero_ws_kernel<<<1024, 256, 0, stream>>>(ws, ws_elems);

    const int nblk = (Nrows + 127) / 128;
    mlp_segsum_kernel<<<nblk, 256, 0, stream>>>(X, y, ids0, ids1, W0, b0, W1, b1,
                                                sum0, sum1, cnt0, cnt1, Nrows);

    head_kernel<<<NQ / 128, 256, 0, stream>>>(sum0, sum1, cnt0, cnt1,
                                              Wm0, bm0, Wv0, bv0,
                                              Wm1, bm1, Wv1, bv1,
                                              eps0, eps1, (float*)d_out);
}
